// RSSM_51754355916917
// MI455X (gfx1250) — matrix operand order repaired and checked by reading, not compiled, not run on hardware
//
#include <hip/hip_runtime.h>

// ---------------------------------------------------------------------------
// RSSM imagine_rollout for gfx1250 (MI455X).
// B=2048, H=64, A=32, S=32, D=200, HID=200.  Out: [B,H,296] f32.
//
// 128 persistent workgroups (256 thr = 8 waves), each owns a 16-row batch
// tile for all 64 steps. GEMMs via v_wmma_f32_16x16x32_f16 (f32 accum).
// Weights pre-packed once per launch into f16 B-fragment layout in d_ws
// (L2-resident); activations in LDS as one concatenated [x|h] buffer,
// double-buffered across steps; every GEMM loop is software-pipelined
// (next A/B fragments loaded before current WMMAs).
// ---------------------------------------------------------------------------

typedef _Float16 v16h __attribute__((ext_vector_type(16)));
typedef _Float16 v8h  __attribute__((ext_vector_type(8)));
typedef float    v8f  __attribute__((ext_vector_type(8)));

#define WMMA_F16(Af, Bf, Cf) \
  __builtin_amdgcn_wmma_f32_16x16x32_f16(false, (Af), false, (Bf), (short)0, (Cf), false, false)

// Fragment table in d_ws (fragment = 512 halves = 32 lanes x 16 halves)
//   G  : [gate 3][jtile 13][kk 9] -> 351   (kk 0-1 from W_ih K=64,
//                                           kk 2-8 from W_hh K=200 pad 224)
//   W1 : [jtile 13][kk 7]         -> 91
//   W2 : [jtile 13][kk 7]         -> 91
//   WM : [jtile 2][kk 7]          -> 14
//   WS : [jtile 2][kk 7]          -> 14
static constexpr int FR_G  = 0;
static constexpr int FR_W1 = 351;
static constexpr int FR_W2 = 442;
static constexpr int FR_WM = 533;
static constexpr int FR_WS = 547;
static constexpr int N_FRAGS = 561;

// ---------------------------------------------------------------------------
// Weight pre-pack: f32 [N,K] row-major -> f16 WMMA B fragments.
// B fragment k order used here (contiguous, same as ldsA): lane n%16 holds column n; lanes 0-15 carry
// K = 0..15 of the fragment contiguously, lanes 16-31 carry K = 16..31.
// ---------------------------------------------------------------------------
__global__ __launch_bounds__(256) void rssm_prep_frags(
    const float* __restrict__ Wih, const float* __restrict__ Whh,
    const float* __restrict__ W1,  const float* __restrict__ W2,
    const float* __restrict__ Wm,  const float* __restrict__ Ws,
    _Float16* __restrict__ frags) {
  int gid  = blockIdx.x * blockDim.x + threadIdx.x;
  int frag = gid >> 5;
  int lane = gid & 31;
  if (frag >= N_FRAGS) return;

  const float* src;
  int row = 0, K = 0, kbase = 0;
  bool valid = true;
  const int n  = lane & 15;
  const int hi = (lane >> 4) << 4;    // lanes 16-31 take K-halves 16..31

  if (frag < FR_W1) {                       // gates: [W_ih | W_hh] per gate
    int g = frag / 117, r = frag % 117, j = r / 9, kk = r % 9;
    int nc = j * 16 + n;  valid = nc < 200; row = g * 200 + nc;
    if (kk < 2) { src = Wih; K = 64;  kbase = kk * 32 + hi; }
    else        { src = Whh; K = 200; kbase = (kk - 2) * 32 + hi; }
  } else if (frag < FR_W2) {                // W1 [200,200]
    int f = frag - FR_W1;
    int j = f / 7, kk = f % 7;
    int nc = j * 16 + n;  valid = nc < 200; row = nc;
    src = W1; K = 200; kbase = kk * 32 + hi;
  } else if (frag < FR_WM) {                // W2 [200,200]
    int f = frag - FR_W2;
    int j = f / 7, kk = f % 7;
    int nc = j * 16 + n;  valid = nc < 200; row = nc;
    src = W2; K = 200; kbase = kk * 32 + hi;
  } else if (frag < FR_WS) {                // Wm [32,200]
    int f = frag - FR_WM;
    int j = f / 7, kk = f % 7;
    row = j * 16 + n;  src = Wm; K = 200; kbase = kk * 32 + hi;
  } else {                                  // Ws [32,200]
    int f = frag - FR_WS;
    int j = f / 7, kk = f % 7;
    row = j * 16 + n;  src = Ws; K = 200; kbase = kk * 32 + hi;
  }

  _Float16* dst = frags + (size_t)frag * 512 + lane * 16;
#pragma unroll
  for (int i = 0; i < 16; ++i) {
    int k = kbase + i;
    float v = (valid && k < K) ? src[(size_t)row * K + k] : 0.0f;
    dst[i] = (_Float16)v;
  }
}

// A-fragment from LDS (contiguous k order used here, same as the B pre-pack): lane row m = lane%16;
// lanes 0-15: K = kk*32 + {0..15}; lanes 16-31: + {16..31}.
static __device__ __forceinline__ v16h ldsA(const _Float16* base, int strideH,
                                            int kk, int lane) {
  const int m  = lane & 15;
  const int k0 = kk * 32 + ((lane >> 4) << 4);
  const _Float16* p = base + m * strideH + k0;
  union { v16h v; v8h h[2]; } u;
  u.h[0] = *(const v8h*)(p);
  u.h[1] = *(const v8h*)(p + 8);
  return u.v;
}

static __device__ __forceinline__ v16h ldB(const _Float16* frags, int frag, int lane) {
  return *((const v16h*)frags + ((size_t)frag * 32 + lane));
}

static __device__ __forceinline__ float sigmoid_f(float x) {
  return 1.0f / (1.0f + __expf(-x));
}
static __device__ __forceinline__ float tanh_f(float x) {
  x = fminf(15.0f, fmaxf(-15.0f, x));
  float e = __expf(2.0f * x);
  return (e - 1.0f) / (e + 1.0f);
}
static __device__ __forceinline__ float elu_f(float x) {
  return x > 0.0f ? x : (__expf(x) - 1.0f);
}

// ---------------------------------------------------------------------------
// Main persistent rollout kernel.
// ---------------------------------------------------------------------------
__global__ __launch_bounds__(256) void rssm_rollout(
    const float* __restrict__ h0, const float* __restrict__ z0,
    const float* __restrict__ actions, const float* __restrict__ eps,
    const float* __restrict__ b_ih, const float* __restrict__ b_hh,
    const float* __restrict__ b1, const float* __restrict__ b2,
    const float* __restrict__ bm, const float* __restrict__ bs,
    const _Float16* __restrict__ frags, float* __restrict__ out) {

  // Activation buffer, double-buffered by step parity:
  // cols 0-31 = z, 32-63 = a, 64-287 = h (200 + zero pad).
  constexpr int ACT_W  = 288;
  constexpr int ACT_SZ = 16 * ACT_W;
  __shared__ _Float16 sAct[2 * ACT_SZ];
  __shared__ float    sH32[16 * 200];   // h, f32 carry
  __shared__ _Float16 sF1[16 * 224];    // MLP hidden 1, f16 (pad = 0)
  __shared__ _Float16 sF2[16 * 224];    // MLP hidden 2, f16 (pad = 0)
  __shared__ float    sMean[16 * 32];
  __shared__ float    sStd[16 * 32];
  __shared__ float    sZ[16 * 32];
  // Cached biases (r/u pre-combined ih+hh)
  __shared__ float    sBr[200], sBu[200], sBni[200], sBnh[200];
  __shared__ float    sB1[200], sB2[200], sBm[32], sBs[32];

  const int tid  = threadIdx.x;
  const int lane = tid & 31;
  const int wid  = tid >> 5;
  const int b0   = blockIdx.x * 16;
  const int col  = lane & 15;       // N column within a 16-wide tile
  const int half = lane >> 4;       // C-layout row offset selector

  // ---- init: zero buffers, cache biases, load h0/z0 into buffer 0 ----
  for (int i = tid; i < 2 * ACT_SZ; i += 256) sAct[i] = (_Float16)0.0f;
  for (int i = tid; i < 16 * 224; i += 256) {
    sF1[i] = (_Float16)0.0f; sF2[i] = (_Float16)0.0f;
  }
  if (tid < 200) {
    sBr[tid]  = b_ih[tid] + b_hh[tid];
    sBu[tid]  = b_ih[200 + tid] + b_hh[200 + tid];
    sBni[tid] = b_ih[400 + tid];
    sBnh[tid] = b_hh[400 + tid];
    sB1[tid]  = b1[tid];
    sB2[tid]  = b2[tid];
  }
  if (tid < 32) { sBm[tid] = bm[tid]; sBs[tid] = bs[tid]; }
  __syncthreads();
  for (int i = tid; i < 16 * 200; i += 256) {
    int m = i / 200, d = i - m * 200;
    float v = h0[(size_t)(b0 + m) * 200 + d];
    sH32[i] = v;
    sAct[m * ACT_W + 64 + d] = (_Float16)v;
  }
  for (int i = tid; i < 16 * 32; i += 256) {
    int m = i >> 5, c = i & 31;
    sAct[m * ACT_W + c] = (_Float16)z0[(size_t)(b0 + m) * 32 + c];
  }

  for (int t = 0; t < 64; ++t) {
    const int p = t & 1, q = p ^ 1;
    const _Float16* actR = sAct + p * ACT_SZ;   // read: [z|a|h_old]
    _Float16*       actW = sAct + q * ACT_SZ;   // write: h_new, z_new

    // ---- stage actions for this step into the read buffer ----
    for (int i = tid; i < 16 * 32; i += 256) {
      int m = i >> 5, c = i & 31;
      sAct[p * ACT_SZ + m * ACT_W + 32 + c] =
          (_Float16)actions[((size_t)(b0 + m) * 64 + t) * 32 + c];
    }
    __syncthreads();

    // ================= GRU gates =================
    // Per j-tile: uniform 9-fragment K loop over concatenated [x|h],
    // software-pipelined (next A/B issued before current WMMAs).
    for (int j = wid; j < 13; j += 8) {
      v8f ar = {}, au = {}, ain = {}, ahn = {};
      const int gr = (0 * 13 + j) * 9, gu = (1 * 13 + j) * 9, gn = (2 * 13 + j) * 9;
      v16h a_c  = ldsA(actR, ACT_W, 0, lane);
      v16h br_c = ldB(frags, FR_G + gr, lane);
      v16h bu_c = ldB(frags, FR_G + gu, lane);
      v16h bn_c = ldB(frags, FR_G + gn, lane);
#pragma unroll
      for (int kk = 0; kk < 9; ++kk) {
        v16h a_n, br_n, bu_n, bn_n;
        if (kk < 8) {
          a_n  = ldsA(actR, ACT_W, kk + 1, lane);
          br_n = ldB(frags, FR_G + gr + kk + 1, lane);
          bu_n = ldB(frags, FR_G + gu + kk + 1, lane);
          bn_n = ldB(frags, FR_G + gn + kk + 1, lane);
        }
        ar = WMMA_F16(a_c, br_c, ar);
        au = WMMA_F16(a_c, bu_c, au);
        if (kk < 2) ain = WMMA_F16(a_c, bn_c, ain);
        else        ahn = WMMA_F16(a_c, bn_c, ahn);
        if (kk < 8) { a_c = a_n; br_c = br_n; bu_c = bu_n; bn_c = bn_n; }
      }
      // epilogue: gate math in registers, C-layout (row = v + 8*half, col)
      const int  nc    = j * 16 + col;
      const bool valid = nc < 200;
      float br = 0.f, bu = 0.f, bni = 0.f, bnh = 0.f;
      if (valid) { br = sBr[nc]; bu = sBu[nc]; bni = sBni[nc]; bnh = sBnh[nc]; }
#pragma unroll
      for (int v = 0; v < 8; ++v) {
        const int m = v + (half << 3);
        float r  = sigmoid_f(ar[v] + br);
        float u  = sigmoid_f(au[v] + bu);
        float nn = tanh_f(ain[v] + bni + r * (ahn[v] + bnh));
        if (valid) {
          float hold = sH32[m * 200 + nc];
          float hnew = (1.0f - u) * nn + u * hold;
          sH32[m * 200 + nc] = hnew;
          actW[m * ACT_W + 64 + nc] = (_Float16)hnew;   // h into write buffer
        }
      }
    }
    __syncthreads();

    // ================= prior MLP layer 1 (reads h_new) =================
    {
      const _Float16* hNew = actW + 64;     // h part of write buffer
      for (int j = wid; j < 13; j += 8) {
        v8f acc = {};
        v16h a_c = ldsA(hNew, ACT_W, 0, lane);
        v16h b_c = ldB(frags, FR_W1 + j * 7, lane);
#pragma unroll
        for (int kk = 0; kk < 7; ++kk) {
          v16h a_n, b_n;
          if (kk < 6) {
            a_n = ldsA(hNew, ACT_W, kk + 1, lane);
            b_n = ldB(frags, FR_W1 + j * 7 + kk + 1, lane);
          }
          acc = WMMA_F16(a_c, b_c, acc);
          if (kk < 6) { a_c = a_n; b_c = b_n; }
        }
        const int  nc    = j * 16 + col;
        const bool valid = nc < 200;
        const float bb   = valid ? sB1[nc] : 0.0f;
#pragma unroll
        for (int v = 0; v < 8; ++v) {
          if (valid) {
            const int m = v + (half << 3);
            sF1[m * 224 + nc] = (_Float16)elu_f(acc[v] + bb);
          }
        }
      }
    }
    __syncthreads();

    // ================= prior MLP layer 2 =================
    for (int j = wid; j < 13; j += 8) {
      v8f acc = {};
      v16h a_c = ldsA(sF1, 224, 0, lane);
      v16h b_c = ldB(frags, FR_W2 + j * 7, lane);
#pragma unroll
      for (int kk = 0; kk < 7; ++kk) {
        v16h a_n, b_n;
        if (kk < 6) {
          a_n = ldsA(sF1, 224, kk + 1, lane);
          b_n = ldB(frags, FR_W2 + j * 7 + kk + 1, lane);
        }
        acc = WMMA_F16(a_c, b_c, acc);
        if (kk < 6) { a_c = a_n; b_c = b_n; }
      }
      const int  nc    = j * 16 + col;
      const bool valid = nc < 200;
      const float bb   = valid ? sB2[nc] : 0.0f;
#pragma unroll
      for (int v = 0; v < 8; ++v) {
        if (valid) {
          const int m = v + (half << 3);
          sF2[m * 224 + nc] = (_Float16)elu_f(acc[v] + bb);
        }
      }
    }
    __syncthreads();

    // ================= mean / std heads =================
    if (wid < 4) {
      const int  jj     = wid & 1;        // which 16-wide slice of S=32
      const bool is_std = wid >= 2;
      const int  fb     = (is_std ? FR_WS : FR_WM) + jj * 7;
      v8f acc = {};
      v16h a_c = ldsA(sF2, 224, 0, lane);
      v16h b_c = ldB(frags, fb, lane);
#pragma unroll
      for (int kk = 0; kk < 7; ++kk) {
        v16h a_n, b_n;
        if (kk < 6) {
          a_n = ldsA(sF2, 224, kk + 1, lane);
          b_n = ldB(frags, fb + kk + 1, lane);
        }
        acc = WMMA_F16(a_c, b_c, acc);
        if (kk < 6) { a_c = a_n; b_c = b_n; }
      }
      const int nc = jj * 16 + col;
#pragma unroll
      for (int v = 0; v < 8; ++v) {
        const int m = v + (half << 3);
        if (!is_std) {
          sMean[m * 32 + nc] = acc[v] + sBm[nc];
        } else {
          float ls = acc[v] + sBs[nc];
          ls = fminf(2.0f, fmaxf(-10.0f, ls));
          sStd[m * 32 + nc] = __expf(ls);
        }
      }
    }
    __syncthreads();

    // ================= reparameterized sample =================
    for (int i = tid; i < 16 * 32; i += 256) {
      int m = i >> 5, c = i & 31;
      float z = sMean[i] +
                sStd[i] * eps[((size_t)(b0 + m) * 64 + t) * 32 + c];
      sZ[i] = z;
      actW[m * ACT_W + c] = (_Float16)z;   // z-part of next step's x
    }
    __syncthreads();

    // ================= coalesced output pass =================
    // out[b0+m, t, :] = concat(h[200], z[32], mean[32], std[32])
    for (int i = tid; i < 16 * 296; i += 256) {
      int m = i / 296, c = i - m * 296;
      float val;
      if (c < 200)       val = sH32[m * 200 + c];
      else if (c < 232)  val = sZ[m * 32 + (c - 200)];
      else if (c < 264)  val = sMean[m * 32 + (c - 232)];
      else               val = sStd[m * 32 + (c - 264)];
      out[((size_t)(b0 + m) * 64 + t) * 296 + c] = val;
    }
    // next iteration's first barrier orders these reads vs. future writes
  }
}

extern "C" void kernel_launch(void* const* d_in, const int* in_sizes, int n_in,
                              void* d_out, int out_size, void* d_ws, size_t ws_size,
                              hipStream_t stream) {
  (void)in_sizes; (void)n_in; (void)out_size; (void)ws_size;
  const float* h0      = (const float*)d_in[0];
  const float* z0      = (const float*)d_in[1];
  const float* actions = (const float*)d_in[2];
  const float* eps     = (const float*)d_in[3];
  const float* W_ih    = (const float*)d_in[4];
  const float* b_ih    = (const float*)d_in[5];
  const float* W_hh    = (const float*)d_in[6];
  const float* b_hh    = (const float*)d_in[7];
  const float* W1      = (const float*)d_in[8];
  const float* b1      = (const float*)d_in[9];
  const float* W2      = (const float*)d_in[10];
  const float* b2      = (const float*)d_in[11];
  const float* Wm      = (const float*)d_in[12];
  const float* bm      = (const float*)d_in[13];
  const float* Ws      = (const float*)d_in[14];
  const float* bs      = (const float*)d_in[15];

  _Float16* frags = (_Float16*)d_ws;   // needs 561*512*2 = 574,464 bytes
  float*    out   = (float*)d_out;

  const int prep_threads = N_FRAGS * 32;
  rssm_prep_frags<<<(prep_threads + 255) / 256, 256, 0, stream>>>(
      W_ih, W_hh, W1, W2, Wm, Ws, frags);

  rssm_rollout<<<128, 256, 0, stream>>>(h0, z0, actions, eps, b_ih, b_hh,
                                        b1, b2, bm, bs, frags, out);
}
